// GATlayer_85237920956844
// MI455X (gfx1250) — matrix operand order repaired and checked by reading, not compiled, not run on hardware
//
#include <hip/hip_runtime.h>
#include <hip/hip_bf16.h>

#define NN 4096
#define FF 256

typedef __bf16 bf16_t;
typedef __attribute__((ext_vector_type(16))) __bf16 v16bf;
typedef __attribute__((ext_vector_type(8)))  __bf16 v8bf;
typedef __attribute__((ext_vector_type(8)))  float  v8f;

// ---------------------------------------------------------------------------
// WMMA fragment loaders (bf16, 16x16x32, wave32)
// A-layout (16-bit A 16x32): lanes 0-15 row M=lane, elems 0-7 = K kb..kb+7,
// elems 8-15 = K kb+16..kb+23, kb = k0 + 8*(lane>=16).
// B-layout (16-bit B 32x16): same K order as A (col N=lane%16, elems 0-7 =
// K kb..kb+7, elems 8-15 = K kb+16..kb+23). B columns are read as contiguous rows of a
// pre-transposed row-major matrix -> all fragments are two 16-byte loads.
// ---------------------------------------------------------------------------
__device__ __forceinline__ v16bf frag_a(const bf16_t* __restrict__ M,
                                        int ld, int r0, int k0, int lane) {
  int row = r0 + (lane & 15);
  int kb  = k0 + ((lane >> 4) << 3);
  const bf16_t* p = M + (size_t)row * ld + kb;
  v8bf lo = *(const v8bf*)(p);
  v8bf hi = *(const v8bf*)(p + 16);
  v16bf r;
#pragma unroll
  for (int j = 0; j < 8; ++j) { r[j] = lo[j]; r[j + 8] = hi[j]; }
  return r;
}

__device__ __forceinline__ v16bf frag_b(const bf16_t* __restrict__ MT,
                                        int ld, int n0, int k0, int lane) {
  int row = n0 + (lane & 15);          // column of B == row of B^T (row-major)
  int kb  = k0 + ((lane >> 4) << 3);
  const bf16_t* p = MT + (size_t)row * ld + kb;
  v8bf lo = *(const v8bf*)(p);
  v8bf hi = *(const v8bf*)(p + 16);
  v16bf r;
#pragma unroll
  for (int j = 0; j < 8; ++j) { r[j] = lo[j]; r[j + 8] = hi[j]; }
  return r;
}

// ---------------------------------------------------------------------------
// 1) Convert X -> bf16, W -> bf16 transposed
// ---------------------------------------------------------------------------
__global__ void k_convert(const float* __restrict__ X, const float* __restrict__ W,
                          bf16_t* __restrict__ Xb, bf16_t* __restrict__ WTb) {
  int i = blockIdx.x * 256 + threadIdx.x;
  if (i < NN * FF) Xb[i] = (bf16_t)X[i];
  if (i < FF * FF) {
    int f = i / FF, o = i % FF;
    WTb[o * FF + f] = (bf16_t)W[i];
  }
}

// ---------------------------------------------------------------------------
// 2) proj = X @ W  (4096x256x256). 16x64 per wave: 1 A-frag / 4 B-frags /
//    4 WMMAs per k-step. Writes proj (f32) and proj^T (bf16).
// ---------------------------------------------------------------------------
__global__ void k_gemm_proj(const bf16_t* __restrict__ Xb, const bf16_t* __restrict__ WTb,
                            float* __restrict__ proj, bf16_t* __restrict__ projTb) {
  int lane = threadIdx.x & 31, wave = threadIdx.x >> 5;
  int tile = blockIdx.x * 8 + wave;            // tilesN = FF/64 = 4
  int m0 = (tile >> 2) * 16;
  int n0 = (tile & 3) * 64;
  v8f acc[4] = {};
#pragma unroll
  for (int k0 = 0; k0 < FF; k0 += 32) {
    v16bf a = frag_a(Xb, FF, m0, k0, lane);
#pragma unroll
    for (int u = 0; u < 4; ++u) {
      v16bf b = frag_b(WTb, FF, n0 + 16 * u, k0, lane);
      acc[u] = __builtin_amdgcn_wmma_f32_16x16x32_bf16(false, a, false, b,
                                                       (short)0, acc[u], false, false);
    }
  }
  int rb = m0 + ((lane >> 4) << 3);
#pragma unroll
  for (int u = 0; u < 4; ++u) {
    int col = n0 + 16 * u + (lane & 15);
#pragma unroll
    for (int r = 0; r < 8; ++r) {
      float v = acc[u][r];
      int row = rb + r;
      proj[row * FF + col]           = v;
      projTb[(size_t)col * NN + row] = (bf16_t)v;
    }
  }
}

// ---------------------------------------------------------------------------
// 3) s_src[i] = proj[i,:]·a_src ; s_tgt[i] = proj[i,:]·a_tgt
// ---------------------------------------------------------------------------
__global__ void k_scores(const float* __restrict__ proj, const float* __restrict__ a_src,
                         const float* __restrict__ a_tgt,
                         float* __restrict__ s_src, float* __restrict__ s_tgt) {
  __shared__ float r1[256], r2[256];
  int row = blockIdx.x, t = threadIdx.x;
  float p = proj[(size_t)row * FF + t];
  r1[t] = p * a_src[t];
  r2[t] = p * a_tgt[t];
  __syncthreads();
  for (int s = 128; s > 0; s >>= 1) {
    if (t < s) { r1[t] += r1[t + s]; r2[t] += r2[t + s]; }
    __syncthreads();
  }
  if (t == 0) { s_src[row] = r1[0]; s_tgt[row] = r2[0]; }
}

// ---------------------------------------------------------------------------
// 4) Row-wise masked softmax of
//    leaky_relu(s_src[i]+s_tgt[j]) - dist*w_d + bond*w_b + mask(degree)
//    -> attn row stored as bf16 (LDS-resident row, single pass over inputs)
// ---------------------------------------------------------------------------
__global__ void k_softmax(const float* __restrict__ s_src, const float* __restrict__ s_tgt,
                          const float* __restrict__ dist, const float* __restrict__ bond,
                          const float* __restrict__ dm,
                          const float* __restrict__ w_dist, const float* __restrict__ w_bond,
                          bf16_t* __restrict__ attnb) {
  __shared__ float z[NN];
  __shared__ float red[256];
  int row = blockIdx.x, t = threadIdx.x;
  size_t base = (size_t)row * NN;
  float wd = w_dist[0], wb = w_bond[0];
  float ss = s_src[row];
  float lmax = -3.4e38f;
  for (int j = t; j < NN; j += 256) {
    float v = ss + s_tgt[j];
    v = v > 0.f ? v : 0.2f * v;                       // leaky_relu
    v += -dist[base + j] * wd + bond[base + j] * wb;  // edge contributions
    float d = dm[base + j];
    v += (d > 0.f) ? d : -1000000.0f;                 // connectivity mask
    z[j] = v;
    lmax = fmaxf(lmax, v);
  }
  red[t] = lmax; __syncthreads();
  for (int s = 128; s > 0; s >>= 1) { if (t < s) red[t] = fmaxf(red[t], red[t + s]); __syncthreads(); }
  float m = red[0]; __syncthreads();
  float lsum = 0.f;
  for (int j = t; j < NN; j += 256) { float e = __expf(z[j] - m); z[j] = e; lsum += e; }
  red[t] = lsum; __syncthreads();
  for (int s = 128; s > 0; s >>= 1) { if (t < s) red[t] += red[t + s]; __syncthreads(); }
  float inv = 1.0f / red[0];
  for (int j = t; j < NN; j += 256) attnb[base + j] = (bf16_t)(z[j] * inv);
}

// ---------------------------------------------------------------------------
// 5) out_pre = attn @ proj  (4096x256, K=4096). 16x64 per wave.
//    Epilogue: torch-faithful transpose scramble (flat idx = j*4096+i) + ELU,
//    writing f32 to d_out[0:N*F) and bf16 copy (o) for the sim GEMM.
// ---------------------------------------------------------------------------
__global__ void k_gemm_out(const bf16_t* __restrict__ attnb, const bf16_t* __restrict__ projTb,
                           float* __restrict__ out_f, bf16_t* __restrict__ ob) {
  int lane = threadIdx.x & 31, wave = threadIdx.x >> 5;
  int tile = blockIdx.x * 8 + wave;            // tilesN = FF/64 = 4
  int m0 = (tile >> 2) * 16;
  int n0 = (tile & 3) * 64;
  v8f acc[4] = {};
  for (int k0 = 0; k0 < NN; k0 += 32) {
    v16bf a = frag_a(attnb, NN, m0, k0, lane);
#pragma unroll
    for (int u = 0; u < 4; ++u) {
      v16bf b = frag_b(projTb, NN, n0 + 16 * u, k0, lane);
      acc[u] = __builtin_amdgcn_wmma_f32_16x16x32_bf16(false, a, false, b,
                                                       (short)0, acc[u], false, false);
    }
  }
  int rb = m0 + ((lane >> 4) << 3);
#pragma unroll
  for (int u = 0; u < 4; ++u) {
    int col = n0 + 16 * u + (lane & 15);
#pragma unroll
    for (int r = 0; r < 8; ++r) {
      int i = rb + r;
      float v = acc[u][r];
      v = v > 0.f ? v : (__expf(v) - 1.0f);           // ELU
      size_t idx = (size_t)col * NN + i;              // transpose scramble
      out_f[idx] = v;
      ob[idx]    = (bf16_t)v;
    }
  }
}

// ---------------------------------------------------------------------------
// 6) sim = o @ o^T (4096x4096, K=256). 32x32 per wave: 2 A x 2 B frags,
//    4 WMMAs per k-step. Fused sigmoid*decay*mask + eps epilogue.
// ---------------------------------------------------------------------------
__global__ void k_gemm_sim(const bf16_t* __restrict__ ob, const float* __restrict__ dist,
                           const float* __restrict__ dm, float* __restrict__ upd) {
  int lane = threadIdx.x & 31, wave = threadIdx.x >> 5;
  int tile = blockIdx.x * 8 + wave;            // tilesN = NN/32 = 128
  int m0 = (tile >> 7) * 32;
  int n0 = (tile & 127) * 32;
  v8f acc[2][2] = {};
#pragma unroll
  for (int k0 = 0; k0 < FF; k0 += 32) {
    v16bf a0 = frag_a(ob, FF, m0,      k0, lane);
    v16bf a1 = frag_a(ob, FF, m0 + 16, k0, lane);
    v16bf b0 = frag_b(ob, FF, n0,      k0, lane);   // cols of o^T == rows of o
    v16bf b1 = frag_b(ob, FF, n0 + 16, k0, lane);
    acc[0][0] = __builtin_amdgcn_wmma_f32_16x16x32_bf16(false, a0, false, b0, (short)0, acc[0][0], false, false);
    acc[0][1] = __builtin_amdgcn_wmma_f32_16x16x32_bf16(false, a0, false, b1, (short)0, acc[0][1], false, false);
    acc[1][0] = __builtin_amdgcn_wmma_f32_16x16x32_bf16(false, a1, false, b0, (short)0, acc[1][0], false, false);
    acc[1][1] = __builtin_amdgcn_wmma_f32_16x16x32_bf16(false, a1, false, b1, (short)0, acc[1][1], false, false);
  }
#pragma unroll
  for (int um = 0; um < 2; ++um) {
    int ib = m0 + 16 * um + ((lane >> 4) << 3);
#pragma unroll
    for (int un = 0; un < 2; ++un) {
      int j = n0 + 16 * un + (lane & 15);
#pragma unroll
      for (int r = 0; r < 8; ++r) {
        int i = ib + r;
        size_t e = (size_t)i * NN + j;
        float sg = 1.0f / (1.0f + __expf(-acc[um][un][r]));
        float mask = dm[e] > 0.f ? 1.0f : 0.0f;
        upd[e] = sg * (-dist[e]) * mask + 1e-6f;
      }
    }
  }
}

// ---------------------------------------------------------------------------
// 7) Per-row layernorm (no affine), in place. Two-pass (mean, then var).
// ---------------------------------------------------------------------------
__global__ void k_layernorm(float* __restrict__ upd) {
  __shared__ float red[256];
  int row = blockIdx.x, t = threadIdx.x;
  size_t base = (size_t)row * NN;
  float vals[16];
  float s = 0.f;
#pragma unroll
  for (int q = 0; q < 16; ++q) { vals[q] = upd[base + t + q * 256]; s += vals[q]; }
  red[t] = s; __syncthreads();
  for (int k = 128; k > 0; k >>= 1) { if (t < k) red[t] += red[t + k]; __syncthreads(); }
  float mean = red[0] * (1.0f / NN); __syncthreads();
  float v = 0.f;
#pragma unroll
  for (int q = 0; q < 16; ++q) { float d = vals[q] - mean; v += d * d; }
  red[t] = v; __syncthreads();
  for (int k = 128; k > 0; k >>= 1) { if (t < k) red[t] += red[t + k]; __syncthreads(); }
  float inv = rsqrtf(red[0] * (1.0f / NN) + 1e-5f);
#pragma unroll
  for (int q = 0; q < 16; ++q) upd[base + t + q * 256] = (vals[q] - mean) * inv;
}

// ---------------------------------------------------------------------------
// 8) upd_out = upd + upd^T
// ---------------------------------------------------------------------------
__global__ void k_symadd(const float* __restrict__ updn, float* __restrict__ out2) {
  size_t idx = (size_t)blockIdx.x * 256 + threadIdx.x;
  int i = (int)(idx >> 12);
  int j = (int)(idx & 4095);
  out2[idx] = updn[(size_t)i * NN + j] + updn[(size_t)j * NN + i];
}

// ---------------------------------------------------------------------------
extern "C" void kernel_launch(void* const* d_in, const int* in_sizes, int n_in,
                              void* d_out, int out_size, void* d_ws, size_t ws_size,
                              hipStream_t stream) {
  (void)in_sizes; (void)n_in; (void)out_size; (void)ws_size;
  const float* nodes  = (const float*)d_in[0];
  const float* degree = (const float*)d_in[1];
  const float* dist   = (const float*)d_in[2];
  const float* bond   = (const float*)d_in[3];
  const float* W      = (const float*)d_in[4];
  const float* a_src  = (const float*)d_in[5];
  const float* a_tgt  = (const float*)d_in[6];
  const float* w_dist = (const float*)d_in[7];
  const float* w_bond = (const float*)d_in[8];
  // d_in[9] = cutoff (unused)

  float* out_f   = (float*)d_out;                 // [N*F] scrambled+ELU output
  float* upd_out = out_f + (size_t)NN * FF;       // [N*N] symmetric update

  // workspace layout (256-byte aligned regions)
  size_t off = 0;
  auto alloc = [&](size_t bytes) { size_t o = off; off = (off + bytes + 255) & ~(size_t)255; return o; };
  char* ws = (char*)d_ws;
  bf16_t* Xb     = (bf16_t*)(ws + alloc((size_t)NN * FF * 2));  // X in bf16
  bf16_t* WTb    = (bf16_t*)(ws + alloc((size_t)FF * FF * 2));  // W^T bf16
  float*  proj   = (float*) (ws + alloc((size_t)NN * FF * 4));  // proj f32
  bf16_t* projTb = (bf16_t*)(ws + alloc((size_t)FF * NN * 2));  // proj^T bf16
  float*  s_src  = (float*) (ws + alloc((size_t)NN * 4));
  float*  s_tgt  = (float*) (ws + alloc((size_t)NN * 4));
  bf16_t* attnb  = (bf16_t*)(ws + alloc((size_t)NN * NN * 2));  // attention bf16
  bf16_t* ob     = (bf16_t*)(ws + alloc((size_t)NN * FF * 2));  // o (post-ELU) bf16
  float*  upd    = (float*) (ws + alloc((size_t)NN * NN * 4));  // update f32

  // 1) dtype conversions
  k_convert<<<(NN * FF + 255) / 256, 256, 0, stream>>>(nodes, W, Xb, WTb);
  // 2) proj = X @ W   (16x64 wave tiles: 256*4 = 1024 tiles / 8 waves)
  k_gemm_proj<<<(NN / 16) * (FF / 64) / 8, 256, 0, stream>>>(Xb, WTb, proj, projTb);
  // 3) per-node attention scores
  k_scores<<<NN, 256, 0, stream>>>(proj, a_src, a_tgt, s_src, s_tgt);
  // 4) masked softmax rows -> bf16 attn
  k_softmax<<<NN, 256, 0, stream>>>(s_src, s_tgt, dist, bond, degree, w_dist, w_bond, attnb);
  // 5) out = ELU(scramble(attn @ proj))   (16x64 wave tiles)
  k_gemm_out<<<(NN / 16) * (FF / 64) / 8, 256, 0, stream>>>(attnb, projTb, out_f, ob);
  // 6) upd_pre = sigmoid(o @ o^T) * decay * mask + eps   (32x32 wave tiles)
  k_gemm_sim<<<(NN / 32) * (NN / 32) / 8, 256, 0, stream>>>(ob, dist, degree, upd);
  // 7) per-row layernorm (in place)
  k_layernorm<<<NN, 256, 0, stream>>>(upd);
  // 8) upd_out = upd + upd^T
  k_symadd<<<(NN * NN) / 256, 256, 0, stream>>>(upd, upd_out);
}
